// TableEncoderBlock_42245298323486
// MI455X (gfx1250) — matrix operand order repaired and checked by reading, not compiled, not run on hardware
//
#include <hip/hip_runtime.h>

// ---------------------------------------------------------------------------
// MI455X / gfx1250. Compute-bound (~550 GFLOP matmul vs ~300MB HBM traffic):
// all matmuls run on v_wmma_f32_16x16x32_bf16 (wave32). GEMM tiles are moved
// global->LDS by the Tensor Data Mover (tensor_load_to_lds, TENSORcnt,
// double-buffered), using TDM pad-insertion to produce a conflict-free
// 56-bf16 LDS row pitch (64B data + 48B pad = 7x16B).
// ---------------------------------------------------------------------------

typedef __attribute__((ext_vector_type(16))) __bf16 v16bf;
typedef __attribute__((ext_vector_type(8)))  float  v8f;
typedef __attribute__((ext_vector_type(4)))  unsigned int u32x4;
typedef __attribute__((ext_vector_type(8)))  int i32x8;
typedef __attribute__((ext_vector_type(4)))  int i32x4;

union FragBF { v16bf v; uint4 q[2]; };
union V8     { uint4 q; __bf16 h[8]; };

__device__ inline v8f wmma_bf16(v16bf a, v16bf b, v8f c) {
  return __builtin_amdgcn_wmma_f32_16x16x32_bf16(false, a, false, b,
                                                 (short)0, c, false, false);
}

// TDM: async-load a 128-row x 32-col bf16 tile (row stride = strideElems)
// into LDS at byte offset ldsOff, inserting 48B pad after each 64B row so the
// LDS pitch is 56 bf16. 2-D descriptor; remaining groups zero.
// This toolchain exposes the 6-arg builtin:
//   (uint32x4 g0, int32x8 g1, int32x4, int32x4, int32x8, i32 cpol).
__device__ inline void tdm_load_tile(unsigned ldsOff, const __bf16* gptr,
                                     unsigned strideElems) {
  unsigned long long ga = (unsigned long long)(size_t)gptr;
  u32x4 g0;
  g0.x = 1u;                                   // count=1 (valid user D#)
  g0.y = ldsOff;                               // lds_addr (bytes)
  g0.z = (unsigned)ga;                         // global_addr[31:0]
  g0.w = (unsigned)(ga >> 32) | (2u << 30);    // global_addr[56:32] | type=2
  i32x8 g1;
  g1[0] = (int)((1u << 16)      // data_size = 2 bytes
              | (1u << 20)      // pad_enable
              | (3u << 22)      // pad_interval: 16 DWORDs (=64B) before pad
              | (11u << 25));   // pad_amount: 12 DWORDs (=48B)
  g1[1] = (int)(32u << 16);     // tensor_dim0[15:0] = 32
  g1[2] = (int)(128u << 16);    // tensor_dim0[31:16]=0 | tensor_dim1[15:0]=128
  g1[3] = (int)(32u << 16);     // tensor_dim1[31:16]=0 | tile_dim0=32
  g1[4] = (int)(128u);          // tile_dim1=128 | tile_dim2=0
  g1[5] = (int)strideElems;     // tensor_dim0_stride[31:0]
  g1[6] = 0;                    // stride[47:32]=0 | dim1_stride lo16=0
  g1[7] = 0;
  i32x4 z4 = {0, 0, 0, 0};
  i32x8 z8 = {0, 0, 0, 0, 0, 0, 0, 0};
  __builtin_amdgcn_tensor_load_to_lds(g0, g1, z4, z4, z8, 0);
}

// Row-index permutations fused into GEMM epilogues.
//  mode 0: identity
//  mode 1: p1=(b*48+k)*48+n  ->  p2=(b*48+n)*48+k      (col-attn out -> row-attn in)
//  mode 2: p1                ->  m = k*384 + b*48 + n  (col_dim final layout)
//  mode 3: p2=(b*48+n)*48+k  ->  m = k*384 + b*48 + n  (row_dim final layout)
__device__ inline int remap_row(int p, int mode) {
  if (mode == 0) return p;
  int b  = p / 2304;
  int x1 = (p / 48) % 48;
  int x0 = p % 48;
  if (mode == 1) return (b * 48 + x0) * 48 + x1;
  if (mode == 2) return x1 * 384 + b * 48 + x0;
  return x0 * 384 + b * 48 + x1;   // mode 3
}

// ---------------------------------------------------------------------------
// Tiled WMMA GEMM: C[M=18432, N=1024] = A[M,KA] * Wt[N,KA]^T (+bias)(tanh)
// Block tile 128x128, 8 waves (2x4), wave tile 64x32 (8 wmma accumulators).
// TDM double-buffered LDS staging: wave 0 issues next-buffer tile DMAs, then
// s_wait_tensorcnt(2) so the current tiles are resident while the next pair
// stays in flight behind the 8 WMMAs per K-step.
// Dynamic LDS layout (byte offsets are TDM lds_addr): A0@0, A1@14336,
// B0@28672, B1@43008 (each 128 rows x 56 bf16 pitch = 14336B). Total 57344B.
// ---------------------------------------------------------------------------
template <bool TANH>
__global__ __launch_bounds__(256) void k_gemm(
    const __bf16* __restrict__ Alo, const __bf16* __restrict__ Ahi, int KA,
    const __bf16* __restrict__ Wt, const float* __restrict__ bias,
    float* __restrict__ dstF, int remapF,
    __bf16* __restrict__ dstB, int remapB) {
  extern __shared__ __bf16 smem[];

  const int tid = threadIdx.x;
  const int wid = tid >> 5, lane = tid & 31;
  const int lhi = lane >> 4, llo = lane & 15;
  const int waveM = wid & 1, waveN = wid >> 1;
  const int m0 = blockIdx.x * 128, n0 = blockIdx.y * 128;
  const int nK = KA >> 5;

  if (wid == 0) {  // prologue: fill buffer 0 (kb = 0, always from Alo)
    tdm_load_tile(0u, Alo + (size_t)m0 * 1024, 1024u);
    tdm_load_tile(28672u, Wt + (size_t)n0 * KA, (unsigned)KA);
  }

  v8f acc[4][2] = {};
  for (int i = 0; i < nK; ++i) {
    const int cur = i & 1;
    const int nxt = cur ^ 1;
    if (wid == 0) {
      if (i + 1 < nK) {
        const int kb = (i + 1) << 5;
        const __bf16* Ab = (kb < 1024) ? Alo : Ahi;
        const int kcol = (kb < 1024) ? kb : (kb - 1024);
        tdm_load_tile((unsigned)nxt * 14336u,
                      Ab + (size_t)m0 * 1024 + kcol, 1024u);
        tdm_load_tile(28672u + (unsigned)nxt * 14336u,
                      Wt + (size_t)n0 * KA + kb, (unsigned)KA);
        __builtin_amdgcn_s_wait_tensorcnt(2);  // current pair landed
      } else {
        __builtin_amdgcn_s_wait_tensorcnt(0);  // last pair landed
      }
    }
    __syncthreads();  // LDS tiles visible to all waves

    const __bf16* sA = smem + cur * 7168;            // buffer cur A tile
    const __bf16* sB = smem + 14336 + cur * 7168;    // buffer cur B tile
    FragBF fa[4], fb[2];
#pragma unroll
    for (int tm = 0; tm < 4; ++tm) {
      const __bf16* s = &sA[(waveM * 64 + tm * 16 + llo) * 56 + lhi * 8];
      fa[tm].q[0] = *(const uint4*)(s);        // K = kb + lhi*8 + 0..7
      fa[tm].q[1] = *(const uint4*)(s + 16);   // K = kb + 16 + lhi*8 + 0..7
    }
#pragma unroll
    for (int tn = 0; tn < 2; ++tn) {
      const __bf16* s = &sB[(waveN * 32 + tn * 16 + llo) * 56 + lhi * 8];
      fb[tn].q[0] = *(const uint4*)(s);        // K = kb + lhi*8 + 0..7
      fb[tn].q[1] = *(const uint4*)(s + 16);   // K = kb + 16 + lhi*8 + 0..7
    }
#pragma unroll
    for (int tm = 0; tm < 4; ++tm)
#pragma unroll
      for (int tn = 0; tn < 2; ++tn)
        acc[tm][tn] = wmma_bf16(fa[tm].v, fb[tn].v, acc[tm][tn]);

    __syncthreads();  // done reading 'cur' before TDM refills it next iter
  }

#pragma unroll
  for (int tm = 0; tm < 4; ++tm) {
#pragma unroll
    for (int tn = 0; tn < 2; ++tn) {
      int N = n0 + waveN * 32 + tn * 16 + llo;
      float bv = bias ? bias[N] : 0.0f;
#pragma unroll
      for (int g = 0; g < 8; ++g) {
        int M = m0 + waveM * 64 + tm * 16 + 8 * lhi + g;
        float x = acc[tm][tn][g] + bv;
        if (TANH) x = tanhf(x);
        if (dstF) dstF[(size_t)remap_row(M, remapF) * 1024 + N] = x;
        if (dstB) dstB[(size_t)remap_row(M, remapB) * 1024 + N] = (__bf16)x;
      }
    }
  }
}

// ---------------------------------------------------------------------------
// WMMA attention: grid (384 seqs, 4), block 64 = 2 waves, one head per wave.
// T=S=48, dh=128, full mask (length==48). S=QK^T (9 wmma tiles, K=128),
// register softmax (shfl_xor across 16-lane halves), P and V^T staged in
// wave-private LDS (pitch 72, K padded 48->64), O=PV (2 x 12 wmma tiles).
// ---------------------------------------------------------------------------
__global__ __launch_bounds__(64) void k_attn(
    const __bf16* __restrict__ Q, const __bf16* __restrict__ K,
    const __bf16* __restrict__ V, __bf16* __restrict__ O) {
  __shared__ __bf16 sP[2][48 * 72];
  __shared__ __bf16 sV[2][128 * 72];
  const int wv = threadIdx.x >> 5, lane = threadIdx.x & 31;
  const int lhi = lane >> 4, llo = lane & 15;
  const int j = blockIdx.x;
  const int h = blockIdx.y * 2 + wv;
  __bf16* P  = sP[wv];
  __bf16* Vt = sV[wv];
  const size_t base = (size_t)j * 48 * 1024 + (size_t)h * 128;
  const uint4 z = make_uint4(0u, 0u, 0u, 0u);

  // Stage V^T (Vt[d][s]) into LDS; zero-pad s=48..63.
  for (int it = 0; it < 24; ++it) {
    int s = it * 2 + lhi;
    int d0 = llo * 8;
    V8 f;
    f.q = *(const uint4*)(V + base + (size_t)s * 1024 + d0);
#pragma unroll
    for (int q = 0; q < 8; ++q) Vt[(d0 + q) * 72 + s] = f.h[q];
  }
  for (int d = lane; d < 128; d += 32) {
    *(uint4*)(&Vt[d * 72 + 48]) = z;
    *(uint4*)(&Vt[d * 72 + 56]) = z;
  }

  // S = Q K^T, fragments straight from global (Q row-major, K rows = B cols).
  v8f acc[3][3] = {};
  for (int kb = 0; kb < 128; kb += 32) {
    FragBF fa[3], fb[3];
#pragma unroll
    for (int t = 0; t < 3; ++t) {
      const __bf16* qp = Q + base + (size_t)(t * 16 + llo) * 1024 + kb + lhi * 8;
      fa[t].q[0] = *(const uint4*)(qp);
      fa[t].q[1] = *(const uint4*)(qp + 16);
      const __bf16* kp = K + base + (size_t)(t * 16 + llo) * 1024 + kb + lhi * 8;
      fb[t].q[0] = *(const uint4*)(kp);
      fb[t].q[1] = *(const uint4*)(kp + 16);
    }
#pragma unroll
    for (int tm = 0; tm < 3; ++tm)
#pragma unroll
      for (int tn = 0; tn < 3; ++tn)
        acc[tm][tn] = wmma_bf16(fa[tm].v, fb[tn].v, acc[tm][tn]);
  }

  // Softmax over s: row M lives in one accumulator slot g across a 16-lane half.
  const float scale = 0.088388347648318447f;  // 1/sqrt(128)
#pragma unroll
  for (int tm = 0; tm < 3; ++tm) {
#pragma unroll
    for (int g = 0; g < 8; ++g) {
      float x0 = acc[tm][0][g] * scale;
      float x1 = acc[tm][1][g] * scale;
      float x2 = acc[tm][2][g] * scale;
      float mx = fmaxf(x0, fmaxf(x1, x2));
      mx = fmaxf(mx, __shfl_xor(mx, 1, 32));
      mx = fmaxf(mx, __shfl_xor(mx, 2, 32));
      mx = fmaxf(mx, __shfl_xor(mx, 4, 32));
      mx = fmaxf(mx, __shfl_xor(mx, 8, 32));
      float e0 = __expf(x0 - mx), e1 = __expf(x1 - mx), e2 = __expf(x2 - mx);
      float sm = e0 + e1 + e2;
      sm += __shfl_xor(sm, 1, 32);
      sm += __shfl_xor(sm, 2, 32);
      sm += __shfl_xor(sm, 4, 32);
      sm += __shfl_xor(sm, 8, 32);
      float inv = 1.0f / sm;
      int M = tm * 16 + 8 * lhi + g;
      P[M * 72 + llo]      = (__bf16)(e0 * inv);
      P[M * 72 + 16 + llo] = (__bf16)(e1 * inv);
      P[M * 72 + 32 + llo] = (__bf16)(e2 * inv);
    }
  }
  for (int r = lane; r < 48; r += 32) {
    *(uint4*)(&P[r * 72 + 48]) = z;
    *(uint4*)(&P[r * 72 + 56]) = z;
  }

  // O = P @ V, two 64-wide N chunks to bound accumulator VGPRs.
  for (int nc = 0; nc < 2; ++nc) {
    v8f oc[3][4] = {};
#pragma unroll
    for (int kb = 0; kb < 64; kb += 32) {
      FragBF fp[3], fv[4];
#pragma unroll
      for (int tm = 0; tm < 3; ++tm) {
        const __bf16* pp = &P[(tm * 16 + llo) * 72 + kb + lhi * 8];
        fp[tm].q[0] = *(const uint4*)(pp);
        fp[tm].q[1] = *(const uint4*)(pp + 16);
      }
#pragma unroll
      for (int tn = 0; tn < 4; ++tn) {
        const __bf16* vp = &Vt[(nc * 64 + tn * 16 + llo) * 72 + kb + lhi * 8];
        fv[tn].q[0] = *(const uint4*)(vp);
        fv[tn].q[1] = *(const uint4*)(vp + 16);
      }
#pragma unroll
      for (int tm = 0; tm < 3; ++tm)
#pragma unroll
        for (int tn = 0; tn < 4; ++tn)
          oc[tm][tn] = wmma_bf16(fp[tm].v, fv[tn].v, oc[tm][tn]);
    }
#pragma unroll
    for (int tm = 0; tm < 3; ++tm)
#pragma unroll
      for (int tn = 0; tn < 4; ++tn)
#pragma unroll
        for (int g = 0; g < 8; ++g) {
          int t = tm * 16 + 8 * lhi + g;
          int d = nc * 64 + tn * 16 + llo;
          O[base + (size_t)t * 1024 + d] = (__bf16)oc[tm][tn][g];
        }
  }
}

// ---- small helper kernels -------------------------------------------------

// fp32 W[K,1024] -> bf16 Wt[1024,K] (transposed so GEMM streams K-contiguous)
__global__ void k_wt(const float* __restrict__ W, __bf16* __restrict__ Wt, int K) {
  int idx = blockIdx.x * 256 + threadIdx.x;
  if (idx >= K * 1024) return;
  int k = idx >> 10, n = idx & 1023;
  Wt[(size_t)n * K + k] = (__bf16)W[idx];
}

// emb(L=2304,B=8,E) -> X1 bf16 in col-attn token order p1=(b*48+k)*48+n
__global__ void k_x1(const float* __restrict__ emb, __bf16* __restrict__ X1) {
  size_t idx = (size_t)blockIdx.x * 256 + threadIdx.x;  // 18432*1024
  int p = (int)(idx >> 10), e = (int)(idx & 1023);
  int b = p / 2304, k = (p / 48) % 48, n = p % 48;
  X1[idx] = (__bf16)emb[((size_t)(n * 48 + k) * 8 + b) * 1024 + e];
}

__global__ void k_f2b(const float* __restrict__ src, __bf16* __restrict__ dst) {
  size_t idx = (size_t)blockIdx.x * 256 + threadIdx.x;
  dst[idx] = (__bf16)src[idx];
}

// score[m] = H_bf16[m,:] . Wmlp2 ; one wave per row, shfl reduction
__global__ __launch_bounds__(256) void k_gemv(const __bf16* __restrict__ H,
                                              const float* __restrict__ w2,
                                              float* __restrict__ s) {
  int wid = threadIdx.x >> 5, lane = threadIdx.x & 31;
  int m = blockIdx.x * 8 + wid;
  const __bf16* hp = H + (size_t)m * 1024;
  float acc = 0.0f;
#pragma unroll 4
  for (int i = 0; i < 32; ++i) {
    int e = i * 32 + lane;
    acc += (float)hp[e] * w2[e];
  }
#pragma unroll
  for (int off = 16; off > 0; off >>= 1) acc += __shfl_xor(acc, off, 32);
  if (lane == 0) s[m] = acc;
}

// softmax([s_row,s_col]) gating + residual home_ply_rep -> bank (m-layout)
__global__ void k_gate(const float* __restrict__ sr, const float* __restrict__ sc,
                       const float* __restrict__ rowF, const float* __restrict__ colF,
                       const float* __restrict__ emb, float* __restrict__ bank) {
  size_t idx = (size_t)blockIdx.x * 256 + threadIdx.x;
  int m = (int)(idx >> 10), e = (int)(idx & 1023);
  float a = sr[m], c = sc[m];
  float mx = fmaxf(a, c);
  float ea = __expf(a - mx), ec = __expf(c - mx);
  float inv = 1.0f / (ea + ec);
  int k = m / 384, i = m % 384, b = i / 48, n = i % 48;
  float home = emb[((size_t)(n * 48 + k) * 8 + b) * 1024 + e];
  bank[idx] = (ea * rowF[idx] + ec * colF[idx]) * inv + home;
}

// out1[n,b,e] = mean_k bank[k*384 + b*48 + n, e]
__global__ void k_mean(const float* __restrict__ bank, float* __restrict__ out1) {
  int idx = blockIdx.x * 256 + threadIdx.x;  // 393216
  int n = idx >> 13, b = (idx >> 10) & 7, e = idx & 1023;
  int basei = b * 48 + n;
  float s = 0.0f;
  for (int k = 0; k < 48; ++k)
    s += bank[((size_t)(k * 384 + basei) << 10) + e];
  out1[idx] = s * (1.0f / 48.0f);
}

// ---------------------------------------------------------------------------
extern "C" void kernel_launch(void* const* d_in, const int* in_sizes, int n_in,
                              void* d_out, int out_size, void* d_ws, size_t ws_size,
                              hipStream_t stream) {
  (void)in_sizes; (void)n_in; (void)out_size; (void)ws_size;
  const float* emb    = (const float*)d_in[0];
  const float* Wq_c   = (const float*)d_in[1];
  const float* bq_c   = (const float*)d_in[2];
  const float* Wk_c   = (const float*)d_in[3];
  const float* bk_c   = (const float*)d_in[4];
  const float* Wv_c   = (const float*)d_in[5];
  const float* bv_c   = (const float*)d_in[6];
  const float* Wo_c   = (const float*)d_in[7];
  const float* bo_c   = (const float*)d_in[8];
  const float* Wq_r   = (const float*)d_in[9];
  const float* bq_r   = (const float*)d_in[10];
  const float* Wk_r   = (const float*)d_in[11];
  const float* bk_r   = (const float*)d_in[12];
  const float* Wv_r   = (const float*)d_in[13];
  const float* bv_r   = (const float*)d_in[14];
  const float* Wo_r   = (const float*)d_in[15];
  const float* bo_r   = (const float*)d_in[16];
  const float* W_gen  = (const float*)d_in[17];
  const float* W_mlp1 = (const float*)d_in[18];
  const float* W_mlp2 = (const float*)d_in[19];

  float* out     = (float*)d_out;
  float* outMean = out;                 // (48,8,1024)           393216
  float* bank    = out + 393216;        // (48,384,1024)       18874368
  float* colF    = out + 19267584;      // col_for_ptr (m-layout fp32)
  float* rowF    = out + 38141952;      // row_for_ptr (m-layout fp32)

  char* p = (char*)d_ws;
  auto alloc = [&](size_t b) { char* r = p; p += (b + 255) & ~(size_t)255; return r; };
  const size_t TOKB = (size_t)18432 * 1024 * sizeof(__bf16);
  __bf16* X1   = (__bf16*)alloc(TOKB);  // col-attn input; reused as attn-out A1
  __bf16* X2   = (__bf16*)alloc(TOKB);  // row-attn input; reused as attn-out A2
  __bf16* Qb   = (__bf16*)alloc(TOKB);
  __bf16* Kb   = (__bf16*)alloc(TOKB);
  __bf16* Vb   = (__bf16*)alloc(TOKB);
  __bf16* rowB = (__bf16*)alloc(TOKB);
  __bf16* colB = (__bf16*)alloc(TOKB);
  __bf16* genB = (__bf16*)alloc(TOKB);
  __bf16* Hb   = (__bf16*)alloc(TOKB);
  const size_t WB = (size_t)1024 * 1024 * sizeof(__bf16);
  __bf16* Wqc_t = (__bf16*)alloc(WB);
  __bf16* Wkc_t = (__bf16*)alloc(WB);
  __bf16* Wvc_t = (__bf16*)alloc(WB);
  __bf16* Woc_t = (__bf16*)alloc(WB);
  __bf16* Wqr_t = (__bf16*)alloc(WB);
  __bf16* Wkr_t = (__bf16*)alloc(WB);
  __bf16* Wvr_t = (__bf16*)alloc(WB);
  __bf16* Wor_t = (__bf16*)alloc(WB);
  __bf16* Wgen_t  = (__bf16*)alloc(2 * WB);
  __bf16* Wmlp1_t = (__bf16*)alloc(2 * WB);
  float* sRow = (float*)alloc(18432 * sizeof(float));
  float* sCol = (float*)alloc(18432 * sizeof(float));

  const dim3 gG(144, 8), bG(256);
  const dim3 gA(384, 4), bA(64);
  const unsigned GEMM_LDS = 57344;  // 2 x (A 14336B + B 14336B)

  // Weight transpose+convert (fp32 -> bf16, [N][K])
  k_wt<<<4096, 256, 0, stream>>>(Wq_c, Wqc_t, 1024);
  k_wt<<<4096, 256, 0, stream>>>(Wk_c, Wkc_t, 1024);
  k_wt<<<4096, 256, 0, stream>>>(Wv_c, Wvc_t, 1024);
  k_wt<<<4096, 256, 0, stream>>>(Wo_c, Woc_t, 1024);
  k_wt<<<4096, 256, 0, stream>>>(Wq_r, Wqr_t, 1024);
  k_wt<<<4096, 256, 0, stream>>>(Wk_r, Wkr_t, 1024);
  k_wt<<<4096, 256, 0, stream>>>(Wv_r, Wvr_t, 1024);
  k_wt<<<4096, 256, 0, stream>>>(Wo_r, Wor_t, 1024);
  k_wt<<<8192, 256, 0, stream>>>(W_gen,  Wgen_t,  2048);
  k_wt<<<8192, 256, 0, stream>>>(W_mlp1, Wmlp1_t, 2048);

  // Gather emb into col-attention token order
  k_x1<<<73728, 256, 0, stream>>>(emb, X1);

  // Stage 1: col MHA
  k_gemm<false><<<gG, bG, GEMM_LDS, stream>>>(X1, nullptr, 1024, Wqc_t, bq_c, nullptr, 0, Qb, 0);
  k_gemm<false><<<gG, bG, GEMM_LDS, stream>>>(X1, nullptr, 1024, Wkc_t, bk_c, nullptr, 0, Kb, 0);
  k_gemm<false><<<gG, bG, GEMM_LDS, stream>>>(X1, nullptr, 1024, Wvc_t, bv_c, nullptr, 0, Vb, 0);
  k_attn<<<gA, bA, 0, stream>>>(Qb, Kb, Vb, X1);
  // O-proj: fp32 col_dim (m-layout, straight into d_out) + bf16 X2 (p2-layout)
  k_gemm<false><<<gG, bG, GEMM_LDS, stream>>>(X1, nullptr, 1024, Woc_t, bo_c, colF, 2, X2, 1);
  k_f2b<<<73728, 256, 0, stream>>>(colF, colB);

  // Stage 2: row MHA
  k_gemm<false><<<gG, bG, GEMM_LDS, stream>>>(X2, nullptr, 1024, Wqr_t, bq_r, nullptr, 0, Qb, 0);
  k_gemm<false><<<gG, bG, GEMM_LDS, stream>>>(X2, nullptr, 1024, Wkr_t, bk_r, nullptr, 0, Kb, 0);
  k_gemm<false><<<gG, bG, GEMM_LDS, stream>>>(X2, nullptr, 1024, Wvr_t, bv_r, nullptr, 0, Vb, 0);
  k_attn<<<gA, bA, 0, stream>>>(Qb, Kb, Vb, X2);
  k_gemm<false><<<gG, bG, GEMM_LDS, stream>>>(X2, nullptr, 1024, Wor_t, bo_r, rowF, 3, nullptr, 0);
  k_f2b<<<73728, 256, 0, stream>>>(rowF, rowB);

  // gen = tanh([row|col] @ W_gen), H = tanh([x|gen] @ W_mlp1), scores = H @ W_mlp2
  k_gemm<true><<<gG, bG, GEMM_LDS, stream>>>(rowB, colB, 2048, Wgen_t,  nullptr, nullptr, 0, genB, 0);
  k_gemm<true><<<gG, bG, GEMM_LDS, stream>>>(rowB, genB, 2048, Wmlp1_t, nullptr, nullptr, 0, Hb, 0);
  k_gemv<<<2304, 256, 0, stream>>>(Hb, W_mlp2, sRow);
  k_gemm<true><<<gG, bG, GEMM_LDS, stream>>>(colB, genB, 2048, Wmlp1_t, nullptr, nullptr, 0, Hb, 0);
  k_gemv<<<2304, 256, 0, stream>>>(Hb, W_mlp2, sCol);

  // Gating + residual, then mean over k
  k_gate<<<73728, 256, 0, stream>>>(sRow, sCol, rowF, colF, emb, bank);
  k_mean<<<1536, 256, 0, stream>>>(bank, outMean);
}
